// DotProductAttentionLayer_46712064311857
// MI455X (gfx1250) — compile-verified, matrix operand order checked by reading, not run on hardware
//
#include <hip/hip_runtime.h>
#include <hip/hip_bf16.h>
#include <math.h>

typedef __bf16 bf16_t;
typedef __attribute__((ext_vector_type(4)))  bf16_t v4bf;
typedef __attribute__((ext_vector_type(8)))  bf16_t v8bf;
typedef __attribute__((ext_vector_type(16))) bf16_t v16bf;
typedef __attribute__((ext_vector_type(8)))  float  v8f;

#define T_DIM 2048
#define B_DIM 16
#define H_DIM 512
#define NROWS (T_DIM * B_DIM)          // 32768 flattened (t,b) rows

// ---------------------------------------------------------------------------
// Build a v16bf A/B fragment for v_wmma_f32_16x16x32_bf16.
// 16-bit A-matrix layout (ISA 7.12.2): lanes 0-15 hold K=0..7 (V0-3) and
// K=16..23 (V4-7); lanes 16-31 hold K=8..15 and K=24..31.  With `p` pointing
// at (row, k0 + (lane>>4)*8) in a row-major bf16 matrix this is two 16-byte
// contiguous loads per lane.  B uses the symmetric layout with lane = column,
// so loading rows of the *transposed* operand uses identical addressing.
// ---------------------------------------------------------------------------
__device__ __forceinline__ v16bf load_frag(const bf16_t* p) {
    v8bf lo = *(const v8bf*)p;          // K chunk +0..7
    v8bf hi = *(const v8bf*)(p + 16);   // K chunk +16..23
    return __builtin_shufflevector(lo, hi, 0,1,2,3,4,5,6,7,8,9,10,11,12,13,14,15);
}

// ---------------------------------------------------------------------------
// Kernel 1: results = concat(inputs, t<2 ? inputs : rand_ctx); Xbf = bf16(inputs)
// ---------------------------------------------------------------------------
__global__ __launch_bounds__(256) void pack_kernel(
    const float* __restrict__ x, const float* __restrict__ rctx,
    float* __restrict__ results, bf16_t* __restrict__ xbf)
{
    int i = blockIdx.x * blockDim.x + threadIdx.x;     // float4 index
    const int n4 = NROWS * H_DIM / 4;                  // 4,194,304
    if (i >= n4) return;
    int e   = i * 4;
    int row = e / H_DIM;                               // t*B + b
    int h   = e - row * H_DIM;
    int t   = row / B_DIM;

    float4 v = ((const float4*)x)[i];
    float4 c = (t < 2) ? v : ((const float4*)rctx)[i];

    float* o = results + (long)row * (2 * H_DIM) + h;
    *(float4*)o            = v;
    *(float4*)(o + H_DIM)  = c;

    v4bf bv = { (bf16_t)v.x, (bf16_t)v.y, (bf16_t)v.z, (bf16_t)v.w };
    *(v4bf*)(xbf + e) = bv;
}

// ---------------------------------------------------------------------------
// Kernel 2: Wt[n][k] = bf16(W[k][n])   (512x512, tiny — perf irrelevant)
// ---------------------------------------------------------------------------
__global__ __launch_bounds__(256) void wtr_kernel(
    const float* __restrict__ W, bf16_t* __restrict__ wt)
{
    int i = blockIdx.x * blockDim.x + threadIdx.x;
    if (i >= H_DIM * H_DIM) return;
    int k = i & (H_DIM - 1);
    int n = i >> 9;
    wt[n * H_DIM + k] = (bf16_t)W[k * H_DIM + n];
}

// ---------------------------------------------------------------------------
// Kernel 3: Qbf = bf16( Xbf @ W )   [32768 x 512] = [32768 x 512][512 x 512]
// One wave -> one 16x64 output strip; A fragments for full K kept in regs.
// ---------------------------------------------------------------------------
__global__ __launch_bounds__(128) void qgemm_kernel(
    const bf16_t* __restrict__ xbf, const bf16_t* __restrict__ wt,
    bf16_t* __restrict__ qbf)
{
    int lane = threadIdx.x & 31;
    int wid  = threadIdx.x >> 5;
    int gw   = blockIdx.x * 4 + wid;            // 16384 waves total
    int m0   = (gw >> 3) * 16;                  // row tile
    int nb   = (gw & 7) * 64;                   // 64-wide N strip
    int half = lane >> 4;
    int mr   = lane & 15;

    const bf16_t* arow = xbf + (long)(m0 + mr) * H_DIM + half * 8;
    v16bf a[16];
#pragma unroll
    for (int kk = 0; kk < 16; ++kk) a[kk] = load_frag(arow + kk * 32);

#pragma unroll
    for (int nt = 0; nt < 4; ++nt) {
        int n0 = nb + nt * 16;
        const bf16_t* brow = wt + (long)(n0 + mr) * H_DIM + half * 8;
        v8f acc = {0.f, 0.f, 0.f, 0.f, 0.f, 0.f, 0.f, 0.f};
#pragma unroll
        for (int kk = 0; kk < 16; ++kk) {
            v16bf bm = load_frag(brow + kk * 32);
            acc = __builtin_amdgcn_wmma_f32_16x16x32_bf16(
                false, a[kk], false, bm, (short)0, acc, false, false);
        }
        // C/D layout: VGPR r -> M = r + 8*half, N = lane&15
        bf16_t* q = qbf + (long)(m0 + half * 8) * H_DIM + n0 + mr;
#pragma unroll
        for (int r = 0; r < 8; ++r) q[(long)r * H_DIM] = (bf16_t)acc[r];
    }
}

// ---------------------------------------------------------------------------
// Kernel 4: per (b, 16-row t-tile): scores = Q_b tile @ X_b^T  (WMMA, f32 acc
// into 16x2048 LDS), then masked causal softmax, stream attn[b][t][:] out.
// 8 waves / block; wave w owns s-tiles w, w+8, ... then softmaxes rows 2w,2w+1.
// ---------------------------------------------------------------------------
__global__ __launch_bounds__(256) void attn_kernel(
    const bf16_t* __restrict__ qbf, const bf16_t* __restrict__ xbf,
    float* __restrict__ attn)
{
    extern __shared__ float sc[];                // 16 * 2048 f32 = 128 KiB
    int b    = blockIdx.x & (B_DIM - 1);
    int t0   = (blockIdx.x >> 4) * 16;
    int lane = threadIdx.x & 31;
    int wid  = threadIdx.x >> 5;
    int half = lane >> 4;
    int mr   = lane & 15;

    // A fragments: Q rows (t0+mr, b), full K=512
    const bf16_t* arow = qbf + ((long)(t0 + mr) * B_DIM + b) * H_DIM + half * 8;
    v16bf a[16];
#pragma unroll
    for (int kk = 0; kk < 16; ++kk) a[kk] = load_frag(arow + kk * 32);

    for (int st = wid; st < T_DIM / 16; st += 8) {
        int s0 = st * 16;
        // B fragment: column n of X_b^T == row (s0+n, b) of X  -> contiguous
        const bf16_t* brow = xbf + ((long)(s0 + mr) * B_DIM + b) * H_DIM + half * 8;
        v8f acc = {0.f, 0.f, 0.f, 0.f, 0.f, 0.f, 0.f, 0.f};
#pragma unroll
        for (int kk = 0; kk < 16; ++kk) {
            v16bf bm = load_frag(brow + kk * 32);
            acc = __builtin_amdgcn_wmma_f32_16x16x32_bf16(
                false, a[kk], false, bm, (short)0, acc, false, false);
        }
#pragma unroll
        for (int r = 0; r < 8; ++r)
            sc[(r + 8 * half) * T_DIM + s0 + mr] = acc[r];
    }
    __syncthreads();

    // Masked softmax: row m valid over s in [0, t) when t >= 2, else all-zero.
#pragma unroll
    for (int rr = 0; rr < 2; ++rr) {
        int m = wid * 2 + rr;
        int t = t0 + m;
        float* orow = attn + ((long)b * T_DIM + t) * T_DIM;
        if (t < 2) {
            for (int s = lane; s < T_DIM; s += 32) orow[s] = 0.f;
            continue;
        }
        int L = t;                                       // s < t
        const float* srow = sc + (long)m * T_DIM;
        float mx = -INFINITY;
        for (int s = lane; s < L; s += 32) mx = fmaxf(mx, srow[s]);
#pragma unroll
        for (int off = 16; off; off >>= 1) mx = fmaxf(mx, __shfl_xor(mx, off, 32));
        float sum = 0.f;
        for (int s = lane; s < L; s += 32) sum += __expf(srow[s] - mx);
#pragma unroll
        for (int off = 16; off; off >>= 1) sum += __shfl_xor(sum, off, 32);
        float inv = 1.f / sum;
        for (int s = lane; s < T_DIM; s += 32)
            orow[s] = (s < L) ? __expf(srow[s] - mx) * inv : 0.f;
    }
}

// ---------------------------------------------------------------------------
extern "C" void kernel_launch(void* const* d_in, const int* in_sizes, int n_in,
                              void* d_out, int out_size, void* d_ws, size_t ws_size,
                              hipStream_t stream)
{
    const float* x    = (const float*)d_in[0];   // inputs   [T,B,H]
    const float* rctx = (const float*)d_in[1];   // rand_ctx [T,B,H]
    const float* W    = (const float*)d_in[2];   // W        [H,H]
    // d_in[3] = attention_width (dead code in reference)

    float* results = (float*)d_out;                                  // [T,B,2H]
    float* attn    = (float*)d_out + (size_t)NROWS * 2 * H_DIM;      // [B,T,T]

    char*   ws  = (char*)d_ws;
    bf16_t* xbf = (bf16_t*)ws;                                       // 32 MiB
    bf16_t* qbf = (bf16_t*)(ws + (size_t)NROWS * H_DIM * 2);         // 32 MiB
    bf16_t* wt  = (bf16_t*)(ws + (size_t)NROWS * H_DIM * 4);         // 0.5 MiB

    // 1) results + bf16 inputs
    {
        int n4 = NROWS * H_DIM / 4;
        pack_kernel<<<(n4 + 255) / 256, 256, 0, stream>>>(x, rctx, results, xbf);
    }
    // 2) W transpose -> bf16
    wtr_kernel<<<(H_DIM * H_DIM + 255) / 256, 256, 0, stream>>>(W, wt);
    // 3) Q = X @ W (bf16 WMMA)
    qgemm_kernel<<<(NROWS / 16) * 8 / 4, 128, 0, stream>>>(xbf, wt, qbf);
    // 4) scores + softmax -> attn (128 KiB dynamic LDS per workgroup)
    attn_kernel<<<B_DIM * (T_DIM / 16), 256, 16 * T_DIM * sizeof(float), stream>>>(
        qbf, xbf, attn);
}
